// KmeansVectorQuantizerVQGAN2_65730179498157
// MI455X (gfx1250) — matrix operand order repaired and checked by reading, not compiled, not run on hardware
//
#include <hip/hip_runtime.h>
#include <hip/hip_bf16.h>
#include <math.h>

typedef __attribute__((ext_vector_type(16))) _Float16 hv16;
typedef __attribute__((ext_vector_type(8)))  _Float16 hv8;
typedef __attribute__((ext_vector_type(4)))  _Float16 hv4;
typedef __attribute__((ext_vector_type(8)))  float    fv8;
typedef __attribute__((ext_vector_type(4)))  float    fv4;

#define NUM_VARS 1024
#define VAR_DIM  64
#define LDS_PITCH 72   // halfs per row: 144B, multiple of 16B, 36-dword bank stride

// ---------------------------------------------------------------------------
// Kernel 1: L2-normalize codebook -> f16, zero the accumulators.
// ---------------------------------------------------------------------------
__global__ void vq_norm_emb(const float* __restrict__ emb,
                            _Float16* __restrict__ embh,
                            float* __restrict__ counts,
                            float* __restrict__ sumsq) {
    int k = blockIdx.x * blockDim.x + threadIdx.x;
    if (k < NUM_VARS) {
        const float* row = emb + k * VAR_DIM;
        float s = 0.0f;
        #pragma unroll
        for (int i = 0; i < VAR_DIM; ++i) s += row[i] * row[i];
        float inv = 1.0f / fmaxf(sqrtf(s), 1e-12f);
        _Float16* dst = embh + k * VAR_DIM;
        #pragma unroll
        for (int i = 0; i < VAR_DIM; ++i) dst[i] = (_Float16)(row[i] * inv);
        counts[k] = 0.0f;
        if (k == 0) sumsq[0] = 0.0f;
    }
}

// ---------------------------------------------------------------------------
// Kernel 2: WMMA nearest-code search + gather + fused stats.
// One wave owns a 16-row tile of x. Codebook (f16) lives in LDS.
// ---------------------------------------------------------------------------
__global__ __launch_bounds__(256) void vq_search(
        const float* __restrict__ x,
        const float* __restrict__ emb,          // unnormalized f32 codebook
        const _Float16* __restrict__ embh,      // normalized f16 codebook
        float* __restrict__ out,                // [N,64] x_out
        float* __restrict__ counts,
        float* __restrict__ sumsq,
        int nrows) {
    __shared__ _Float16 sEmb[NUM_VARS * LDS_PITCH];
    __shared__ int sIdx[8 * 16];

    const int tid  = threadIdx.x;
    const int lane = tid & 31;
    const int wave = tid >> 5;
    const int r    = lane & 15;
    const int hi   = lane >> 4;

    // Stage codebook into LDS, 8 bytes (4 halfs) per transfer.
    {
        const hv4* src = (const hv4*)embh;            // 1024*16 chunks
        for (int ch = tid; ch < NUM_VARS * 16; ch += 256) {
            int row = ch >> 4;
            int cp  = (ch & 15) * 4;
            *(hv4*)(&sEmb[row * LDS_PITCH + cp]) = src[ch];
        }
    }
    __syncthreads();

    // Per-lane base for B-fragment reads (column code = n0 + r).
    const _Float16* bbase = &sEmb[r * LDS_PITCH + 8 * hi];

    const int tiles  = nrows >> 4;
    const int totalW = gridDim.x * 8;
    for (int t = blockIdx.x * 8 + wave; t < tiles; t += totalW) {
        const int m0 = t << 4;
        // Prefetch next tile's A rows while this one computes.
        if (t + totalW < tiles)
            __builtin_prefetch(x + (size_t)((t + totalW) * 16 + r) * VAR_DIM, 0, 1);

        // ---- A fragments (ISA 16-bit A 16x32 layout) ----
        const float* arow = x + (size_t)(m0 + r) * VAR_DIM;
        fv4 f0 = *(const fv4*)(arow + 8 * hi);
        fv4 f1 = *(const fv4*)(arow + 8 * hi + 4);
        fv4 f2 = *(const fv4*)(arow + 16 + 8 * hi);
        fv4 f3 = *(const fv4*)(arow + 16 + 8 * hi + 4);
        fv4 f4 = *(const fv4*)(arow + 32 + 8 * hi);
        fv4 f5 = *(const fv4*)(arow + 32 + 8 * hi + 4);
        fv4 f6 = *(const fv4*)(arow + 48 + 8 * hi);
        fv4 f7 = *(const fv4*)(arow + 48 + 8 * hi + 4);
        hv16 a0, a1;
        #pragma unroll
        for (int j = 0; j < 4; ++j) {
            a0[j]      = (_Float16)f0[j];
            a0[4 + j]  = (_Float16)f1[j];
            a0[8 + j]  = (_Float16)f2[j];
            a0[12 + j] = (_Float16)f3[j];
            a1[j]      = (_Float16)f4[j];
            a1[4 + j]  = (_Float16)f5[j];
            a1[8 + j]  = (_Float16)f6[j];
            a1[12 + j] = (_Float16)f7[j];
        }

        float best[8];
        int   bidx[8];
        #pragma unroll
        for (int v = 0; v < 8; ++v) { best[v] = -3.402823466e38f; bidx[v] = 0; }

        // B-fragment loader for code tile n0 (ISA 16-bit B 32x16 layout).
        auto loadB = [&](int n0, hv16& b0, hv16& b1) {
            const _Float16* bp = bbase + n0 * LDS_PITCH;
            hv8 l0 = *(const hv8*)(bp);
            hv8 l1 = *(const hv8*)(bp + 16);
            hv8 l2 = *(const hv8*)(bp + 32);
            hv8 l3 = *(const hv8*)(bp + 48);
            #pragma unroll
            for (int j = 0; j < 8; ++j) {
                b0[j] = l0[j]; b0[8 + j] = l1[j];
                b1[j] = l2[j]; b1[8 + j] = l3[j];
            }
        };

        // ---- sweep all 1024 codes, software-pipelined by one tile so the
        //      argmax VALU work fills the WMMA->VALU hazard slots ----
        hv16 b0, b1;
        loadB(0, b0, b1);
        fv8 cz = {};
        fv8 cp = __builtin_amdgcn_wmma_f32_16x16x32_f16(false, a0, false, b0,
                                                        (short)0, cz, false, false);
        cp = __builtin_amdgcn_wmma_f32_16x16x32_f16(false, a1, false, b1,
                                                    (short)0, cp, false, false);
        for (int n0 = 16; n0 < NUM_VARS; n0 += 16) {
            loadB(n0, b0, b1);
            fv8 c = __builtin_amdgcn_wmma_f32_16x16x32_f16(false, a0, false, b0,
                                                           (short)0, cz, false, false);
            c = __builtin_amdgcn_wmma_f32_16x16x32_f16(false, a1, false, b1,
                                                       (short)0, c, false, false);
            const int col = (n0 - 16) + r;
            #pragma unroll
            for (int v = 0; v < 8; ++v) {
                bool better = cp[v] > best[v];
                best[v] = better ? cp[v] : best[v];
                bidx[v] = better ? col   : bidx[v];
            }
            cp = c;
        }
        {
            const int col = (NUM_VARS - 16) + r;
            #pragma unroll
            for (int v = 0; v < 8; ++v) {
                bool better = cp[v] > best[v];
                best[v] = better ? cp[v] : best[v];
                bidx[v] = better ? col   : bidx[v];
            }
        }

        // ---- argmax across the 16 lanes of each half (first index on ties),
        //      branchless: v_cmp + v_cndmask only, no EXEC churn ----
        #pragma unroll
        for (int off = 1; off < 16; off <<= 1) {
            #pragma unroll
            for (int v = 0; v < 8; ++v) {
                float ov = __shfl_xor(best[v], off, 32);
                int   oi = __shfl_xor(bidx[v], off, 32);
                bool take = (ov > best[v]) | ((ov == best[v]) & (oi < bidx[v]));
                best[v] = take ? ov : best[v];
                bidx[v] = take ? oi : bidx[v];
            }
        }
        if (r == 0) {
            #pragma unroll
            for (int v = 0; v < 8; ++v) {
                sIdx[wave * 16 + 8 * hi + v] = bidx[v];
                atomicAdd(&counts[bidx[v]], 1.0f);
            }
        }

        // ---- gather unnormalized codes, write x_out, fused commitment loss ----
        float acc = 0.0f;
        #pragma unroll 4
        for (int e = lane; e < 16 * VAR_DIM; e += 32) {
            int m  = e >> 6;
            int cc = e & 63;
            int id = sIdx[wave * 16 + m];
            float q  = emb[(size_t)id * VAR_DIM + cc];
            float xv = x[(size_t)(m0 + m) * VAR_DIM + cc];
            out[(size_t)(m0 + m) * VAR_DIM + cc] = q;
            float d = xv - q;
            acc += d * d;
        }
        #pragma unroll
        for (int off = 1; off < 32; off <<= 1) acc += __shfl_xor(acc, off, 32);
        if (lane == 0) atomicAdd(sumsq, acc);
    }
}

// ---------------------------------------------------------------------------
// Kernel 3: perplexity + kmloss + copy codeuse.
// ---------------------------------------------------------------------------
__global__ void vq_finalize(const float* __restrict__ counts,
                            const float* __restrict__ sumsq,
                            float* __restrict__ out,
                            int nrows) {
    __shared__ float red[256];
    const size_t base = (size_t)nrows * VAR_DIM;
    const float invN = 1.0f / (float)nrows;
    int t = threadIdx.x;
    float s = 0.0f;
    for (int k = t; k < NUM_VARS; k += 256) {
        float c = counts[k];
        out[base + 2 + k] = c;                 // codeuse
        float p = c * invN;
        s += p * logf(p + 1e-7f);
    }
    red[t] = s;
    __syncthreads();
    for (int w = 128; w > 0; w >>= 1) {
        if (t < w) red[t] += red[t + w];
        __syncthreads();
    }
    if (t == 0) {
        out[base + 1] = expf(-red[0]);                                   // perplexity
        out[base]     = 0.25f * sumsq[0] / ((float)nrows * VAR_DIM);     // kmloss
    }
}

// ---------------------------------------------------------------------------
extern "C" void kernel_launch(void* const* d_in, const int* in_sizes, int n_in,
                              void* d_out, int out_size, void* d_ws, size_t ws_size,
                              hipStream_t stream) {
    const float* x   = (const float*)d_in[0];
    const float* emb = (const float*)d_in[1];
    float* out = (float*)d_out;
    const int nrows = in_sizes[0] / VAR_DIM;   // 131072

    _Float16* embh = (_Float16*)d_ws;
    float* counts  = (float*)((char*)d_ws + NUM_VARS * VAR_DIM * sizeof(_Float16));
    float* sumsq   = (float*)((char*)counts + NUM_VARS * sizeof(float));

    vq_norm_emb<<<4, 256, 0, stream>>>(emb, embh, counts, sumsq);
    vq_search<<<512, 256, 0, stream>>>(x, emb, embh, out, counts, sumsq, nrows);
    vq_finalize<<<1, 256, 0, stream>>>(counts, sumsq, out, nrows);
}
